// SupConLoss_52759378264228
// MI455X (gfx1250) — matrix operand order repaired and checked by reading, not compiled, not run on hardware
//
#include <hip/hip_runtime.h>

typedef __attribute__((ext_vector_type(16))) _Float16 v16h;
typedef __attribute__((ext_vector_type(16), aligned(16))) _Float16 v16hu; // 16B-aligned LDS view
typedef __attribute__((ext_vector_type(8)))  _Float16 v8h;
typedef __attribute__((ext_vector_type(4)))  _Float16 v4h;
typedef __attribute__((ext_vector_type(8)))  float    v8f;
typedef __attribute__((ext_vector_type(4)))  int      v4i;

#define SC_EPS   1e-12f
#define SC_LN2   0.6931471805599453f
#define SC_SCALE 4.53981604f   // sqrt((1/0.07) * log2(e)); WMMA out = sim*log2e

constexpr int D          = 128;   // feature dim (fixed by reference)
constexpr int CSPLIT     = 16;    // column phases (occupancy)
constexpr int CHUNK_COLS = 128;   // sim-columns staged per LDS chunk
constexpr int LDSP       = D + 8; // padded pitch (halves): 272B row -> conflict-free

#if __has_builtin(__builtin_amdgcn_global_load_async_to_lds_b128)
#define SC_ASYNC 1
#else
#define SC_ASYNC 0
#endif

#if __has_builtin(__builtin_amdgcn_exp2f)
#define EXP2F(x) __builtin_amdgcn_exp2f(x)
#else
#define EXP2F(x) __expf((x) * SC_LN2)
#endif

#if SC_ASYNC
__device__ __forceinline__ void sc_async16(const _Float16* g, _Float16* l) {
    __builtin_amdgcn_global_load_async_to_lds_b128((v4i*)g, (v4i*)l, 0, 0);
}
#if __has_builtin(__builtin_amdgcn_s_wait_asynccnt)
#define SC_WAIT_ASYNC(n) __builtin_amdgcn_s_wait_asynccnt(n)
#else
#define SC_WAIT_ASYNC(n) asm volatile("s_wait_asynccnt %0" ::"i"(n) : "memory")
#endif
#else
#define SC_WAIT_ASYNC(n)
#endif

// ---------------------------------------------------------------------------
// Kernel 0: zero stats (denom,pos), label histogram bins, and the output.
// ---------------------------------------------------------------------------
__global__ void supcon_prep(float* __restrict__ stats, int nStats,
                            int* __restrict__ hist, float* __restrict__ out) {
    int idx = blockIdx.x * blockDim.x + threadIdx.x;
    if (idx < nStats) stats[idx] = 0.0f;
    if (idx < 16) hist[idx] = 0;
    if (idx == 0) out[0] = 0.0f;
}

// ---------------------------------------------------------------------------
// Kernel 1: L2-normalize rows, fold the sqrt(TEMP_INV*log2e) scale into the
// fp16 copy; also convert labels -> int32 and build the 16-bin histogram.
// ---------------------------------------------------------------------------
__global__ __launch_bounds__(256)
void supcon_normalize(const float* __restrict__ feat, _Float16* __restrict__ fh,
                      const long long* __restrict__ labels64,
                      int* __restrict__ labels32, int* __restrict__ hist, int N) {
    int tid = blockIdx.x * blockDim.x + threadIdx.x;
    if (tid < N) {
        int lab = (int)labels64[tid];
        labels32[tid] = lab;
        atomicAdd(&hist[lab], 1);
    }

    int wave = threadIdx.x >> 5;
    int lane = threadIdx.x & 31;
    int row  = blockIdx.x * 8 + wave;
    if (row >= N) return;

    const float4 v = ((const float4*)(feat + (size_t)row * D))[lane];
    float ss = v.x * v.x + v.y * v.y + v.z * v.z + v.w * v.w;
#pragma unroll
    for (int m = 1; m < 32; m <<= 1) ss += __shfl_xor(ss, m, 32);

    float inv = SC_SCALE / fmaxf(sqrtf(ss), SC_EPS);

    v4h o;
    o[0] = (_Float16)(v.x * inv);
    o[1] = (_Float16)(v.y * inv);
    o[2] = (_Float16)(v.z * inv);
    o[3] = (_Float16)(v.w * inv);
    ((v4h*)(fh + (size_t)row * D))[lane] = o;
}

// ---------------------------------------------------------------------------
// Kernel 2: fused f*f^T (WMMA f16->f32, output already in log2 units) +
// branchless row-stat accumulation. Wave owns two 16-row tiles (M=32);
// 8 waves share B via double-buffered LDS filled with async-to-LDS copies.
// ---------------------------------------------------------------------------
__global__ __launch_bounds__(256)
void supcon_main(const _Float16* __restrict__ fh,
                 const int* __restrict__ labels32,
                 float* __restrict__ denomArr,
                 float* __restrict__ posArr,
                 int N) {
    __shared__ _Float16 Bs[2][CHUNK_COLS * LDSP];  // 2 x 34 KB
    __shared__ int      Ls[2][CHUNK_COLS];         // staged labels

    const int lane  = threadIdx.x & 31;
    // Wave index as a proven-uniform SGPR: makes the diagonal guards scalar.
    const int waveU = __builtin_amdgcn_readfirstlane(threadIdx.x >> 5);
    const int h     = lane >> 4;
    const int l15   = lane & 15;

    const int wrow0 = blockIdx.x * 256 + waveU * 32;   // SGPR-resident
    const int wrow1 = wrow0 + 16;

    // ---- A fragments (16-bit A 16x32 layout), both row groups -------------
    v16h A0[4], A1[4];
    {
        const _Float16* ap0 = fh + (size_t)(wrow0 + l15) * D;
        const _Float16* ap1 = fh + (size_t)(wrow1 + l15) * D;
#pragma unroll
        for (int kc = 0; kc < 4; ++kc) {
            v8h lo = *(const v8h*)(ap0 + kc * 32 + h * 8);
            v8h hi = *(const v8h*)(ap0 + kc * 32 + 16 + h * 8);
            A0[kc] = __builtin_shufflevector(lo, hi, 0, 1, 2, 3, 4, 5, 6, 7,
                                             8, 9, 10, 11, 12, 13, 14, 15);
            v8h lo1 = *(const v8h*)(ap1 + kc * 32 + h * 8);
            v8h hi1 = *(const v8h*)(ap1 + kc * 32 + 16 + h * 8);
            A1[kc] = __builtin_shufflevector(lo1, hi1, 0, 1, 2, 3, 4, 5, 6, 7,
                                             8, 9, 10, 11, 12, 13, 14, 15);
        }
    }

    int labi0[8], labi1[8];
#pragma unroll
    for (int r = 0; r < 8; ++r) {
        labi0[r] = labels32[wrow0 + 8 * h + r];
        labi1[r] = labels32[wrow1 + 8 * h + r];
    }

    float den0[8], pos0[8], den1[8], pos1[8];
#pragma unroll
    for (int r = 0; r < 8; ++r) {
        den0[r] = pos0[r] = 0.0f;
        den1[r] = pos1[r] = 0.0f;
    }

    const int colsPerPhase = N / CSPLIT;
    const int phaseBeg     = blockIdx.y * colsPerPhase;
    const int nChunks      = colsPerPhase / CHUNK_COLS;

    const int ldRow  = threadIdx.x >> 1;   // 0..127
    const int ldHalf = threadIdx.x & 1;    // 0/1 -> 64-half slice

    auto stage = [&](int colChunk, int buf) {
        const _Float16* src = fh + (size_t)(colChunk + ldRow) * D + ldHalf * 64;
        _Float16* dstg = &Bs[buf][ldRow * LDSP + ldHalf * 64];
#if SC_ASYNC
#pragma unroll
        for (int q = 0; q < 8; ++q)
            sc_async16(src + q * 8, dstg + q * 8);
#else
#pragma unroll
        for (int q = 0; q < 8; ++q)
            ((float4*)dstg)[q] = ((const float4*)src)[q];
#endif
        if (threadIdx.x < CHUNK_COLS)
            Ls[buf][threadIdx.x] = labels32[colChunk + threadIdx.x];
    };

    stage(phaseBeg, 0);

    for (int c = 0; c < nChunks; ++c) {
        const int colChunk = phaseBeg + c * CHUNK_COLS;
        const int buf = c & 1;

        if (c + 1 < nChunks) {
            stage(colChunk + CHUNK_COLS, buf ^ 1);
            SC_WAIT_ASYNC(8);   // current chunk's 8 async ops done (in-order)
        } else {
            SC_WAIT_ASYNC(0);
        }
        __syncthreads();

#pragma unroll 2
        for (int tt = 0; tt < CHUNK_COLS / 16; ++tt) {
            const int colBase = colChunk + tt * 16;
            // 16-bit B 32x16 layout: lane(n=l15,h): elem s<8 = fh[col+n][K0+8h+s], s>=8 = fh[col+n][K0+16+8h+s-8]
            const _Float16* bp = &Bs[buf][(tt * 16 + l15) * LDSP + h * 8];

            v8f c0 = {}, c1 = {};
#pragma unroll
            for (int kc = 0; kc < 4; ++kc) {
                v8h blo = *(const v8h*)(bp + kc * 32);
                v8h bhi = *(const v8h*)(bp + kc * 32 + 16);
                v16h B = __builtin_shufflevector(blo, bhi, 0, 1, 2, 3, 4, 5, 6, 7,
                                                 8, 9, 10, 11, 12, 13, 14, 15);
                c0 = __builtin_amdgcn_wmma_f32_16x16x32_f16(
                         false, A0[kc], false, B, (short)0, c0, false, false);
                c1 = __builtin_amdgcn_wmma_f32_16x16x32_f16(
                         false, A1[kc], false, B, (short)0, c1, false, false);
            }

            const int labj = Ls[buf][tt * 16 + l15];

            // Branchless: c already equals sim*log2e -> v_exp directly.
#pragma unroll
            for (int r = 0; r < 8; ++r) {
                float s = c0[r];
                den0[r] += EXP2F(s);
                bool m = (labj == labi0[r]);
                pos0[r] += m ? s : 0.0f;
            }
#pragma unroll
            for (int r = 0; r < 8; ++r) {
                float s = c1[r];
                den1[r] += EXP2F(s);
                bool m = (labj == labi1[r]);
                pos1[r] += m ? s : 0.0f;
            }

            // Scalar-uniform diagonal correction (hits once per wave total).
            if (colBase == wrow0) {
#pragma unroll
                for (int r = 0; r < 8; ++r) {
                    bool d  = (l15 == 8 * h + r);
                    float s = c0[r];
                    den0[r] -= d ? EXP2F(s) : 0.0f;
                    pos0[r] -= d ? s : 0.0f;
                }
            }
            if (colBase == wrow1) {
#pragma unroll
                for (int r = 0; r < 8; ++r) {
                    bool d  = (l15 == 8 * h + r);
                    float s = c1[r];
                    den1[r] -= d ? EXP2F(s) : 0.0f;
                    pos1[r] -= d ? s : 0.0f;
                }
            }
        }
        __syncthreads();
    }

    // ---- Segmented 16-lane reductions, fold into per-row arrays -----------
#pragma unroll
    for (int r = 0; r < 8; ++r) {
#pragma unroll
        for (int m = 1; m < 16; m <<= 1) {
            den0[r] += __shfl_xor(den0[r], m, 32);
            pos0[r] += __shfl_xor(pos0[r], m, 32);
            den1[r] += __shfl_xor(den1[r], m, 32);
            pos1[r] += __shfl_xor(pos1[r], m, 32);
        }
        if (l15 == r) {
            const int i0 = wrow0 + 8 * h + r;
            atomicAdd(&denomArr[i0], den0[r]);
            atomicAdd(&posArr[i0],   pos0[r]);
            const int i1 = wrow1 + 8 * h + r;
            atomicAdd(&denomArr[i1], den1[r]);
            atomicAdd(&posArr[i1],   pos1[r]);
        }
    }
}

// ---------------------------------------------------------------------------
// Kernel 3: per-row loss; positive count comes from the label histogram.
// ---------------------------------------------------------------------------
__global__ __launch_bounds__(256)
void supcon_finalize(const float* __restrict__ denomArr,
                     const float* __restrict__ posArr,
                     const int* __restrict__ labels32,
                     const int* __restrict__ hist,
                     float* __restrict__ out, int N) {
    int i = blockIdx.x * blockDim.x + threadIdx.x;
    float contrib = 0.0f;
    if (i < N) {
        float cntf  = (float)(hist[labels32[i]] - 1);
        float posN  = posArr[i] * SC_LN2;  // back to natural-log units
        float loss  = -(posN - cntf * __logf(denomArr[i] + SC_EPS)) /
                      (cntf + SC_EPS);
        contrib = loss / (float)N;
    }
#pragma unroll
    for (int m = 1; m < 32; m <<= 1) contrib += __shfl_xor(contrib, m, 32);
    if ((threadIdx.x & 31) == 0) atomicAdd(out, contrib);
}

// ---------------------------------------------------------------------------
extern "C" void kernel_launch(void* const* d_in, const int* in_sizes, int n_in,
                              void* d_out, int out_size, void* d_ws, size_t ws_size,
                              hipStream_t stream) {
    (void)n_in; (void)out_size; (void)ws_size;
    const float*     feat     = (const float*)d_in[0];
    const long long* labels64 = (const long long*)d_in[1];
    float*           out      = (float*)d_out;
    const int N = in_sizes[1];            // 8192

    // Workspace: [fp16 scaled-normalized features | denom | pos | labels32 | hist]
    _Float16* fh    = (_Float16*)d_ws;
    float*    stats = (float*)((char*)d_ws + (size_t)N * D * sizeof(_Float16));
    float* denomArr = stats;
    float* posArr   = stats + N;
    int*   labels32 = (int*)(stats + 2 * N);
    int*   hist     = labels32 + N;

    supcon_prep<<<(2 * N + 255) / 256, 256, 0, stream>>>(stats, 2 * N, hist, out);
    supcon_normalize<<<N / 8, 256, 0, stream>>>(feat, fh, labels64, labels32, hist, N);
    dim3 grid(N / 256, CSPLIT);
    supcon_main<<<grid, 256, 0, stream>>>(fh, labels32, denomArr, posArr, N);
    supcon_finalize<<<(N + 255) / 256, 256, 0, stream>>>(denomArr, posArr,
                                                         labels32, hist, out, N);
}
